// RoadDiffusion_89867895701906
// MI455X (gfx1250) — matrix operand order repaired and checked by reading, not compiled, not run on hardware
//
#include <hip/hip_runtime.h>
#include <hip/hip_bf16.h>
#include <cstdint>
#include <cstddef>

typedef __attribute__((ext_vector_type(16))) _Float16 v16h;
typedef __attribute__((ext_vector_type(8)))  _Float16 v8h;
typedef __attribute__((ext_vector_type(8)))  float    v8f;

static constexpr int T_  = 12;
static constexpr int NR_ = 512;
static constexpr int NL_ = 1024;
static constexpr int B_  = 8;
static constexpr int M_  = B_ * NR_;   // 4096 attention rows
static constexpr float SLOPE = 0.2f;

// ---- CDNA5 async copy (global -> LDS), tracked by ASYNCcnt ---------------
// loff = byte offset within workgroup LDS (low 32 bits of generic pointer);
// gptr = 64-bit global address. Per-lane transfer, EXEC all-ones here.
#define ASYNC_LDS_B32(loff, gptr)                                          \
  asm volatile("global_load_async_to_lds_b32 %0, %1, off" ::"v"(loff),     \
               "v"(gptr)                                                    \
               : "memory")
#define ASYNC_LDS_B128(loff, gptr)                                         \
  asm volatile("global_load_async_to_lds_b128 %0, %1, off" ::"v"(loff),    \
               "v"(gptr)                                                    \
               : "memory")
#define WAIT_ASYNC(n) asm volatile("s_wait_asynccnt " #n ::: "memory")

__device__ __forceinline__ unsigned lds_off(const void* p) {
  return (unsigned)(size_t)p;  // generic LDS ptr: aperture in hi32, offset in lo32
}

// ---------------------------------------------------------------- helpers
__device__ __forceinline__ float lrelu(float x) { return x > 0.f ? x : SLOPE * x; }

__device__ __forceinline__ unsigned pcg_hash(unsigned v) {
  v = v * 747796405u + 2891336453u;
  unsigned w = ((v >> ((v >> 28u) + 4u)) ^ v) * 277803737u;
  return (w >> 22u) ^ w;
}
__device__ __forceinline__ float hash_gauss(unsigned seed) {
  unsigned a = pcg_hash(seed), b = pcg_hash(seed ^ 0x9e3779b9u);
  float u1 = (a >> 8) * (1.0f / 16777216.0f) + 1e-7f;
  float u2 = (b >> 8) * (1.0f / 16777216.0f);
  return sqrtf(-2.f * __logf(u1)) * __cosf(6.28318530718f * u2);
}

__device__ __forceinline__ v8f wmma_f16(v16h a, v16h b, v8f c) {
  // D = A(16x32 f16) * B(32x16 f16) + C(16x16 f32)
  return __builtin_amdgcn_wmma_f32_16x16x32_f16(
      /*neg_a=*/false, a, /*neg_b=*/false, b,
      /*c_mod=*/(short)0, c, /*reuse_a=*/false, /*reuse_b=*/false);
}

// ------------------------------------------------- RevIN stats + normalize
__global__ void k_revin(const float* __restrict__ x, const float* __restrict__ rw,
                        const float* __restrict__ rb, float* __restrict__ meanb,
                        float* __restrict__ stdb, float* __restrict__ xn) {
  __shared__ float sm[T_], sq[T_], smean[T_], sstd[T_];
  const int b = blockIdx.x, tid = threadIdx.x;
  if (tid < T_) { sm[tid] = 0.f; sq[tid] = 0.f; }
  __syncthreads();
  float ls[T_], lq[T_];
#pragma unroll
  for (int t = 0; t < T_; ++t) { ls[t] = 0.f; lq[t] = 0.f; }
  for (int n = tid; n < NR_; n += blockDim.x) {
    const float* row = x + ((size_t)b * NR_ + n) * T_;
#pragma unroll
    for (int t = 0; t < T_; ++t) { float v = row[t]; ls[t] += v; lq[t] += v * v; }
  }
#pragma unroll
  for (int t = 0; t < T_; ++t) { atomicAdd(&sm[t], ls[t]); atomicAdd(&sq[t], lq[t]); }
  __syncthreads();
  if (tid < T_) {
    float mu  = sm[tid] * (1.0f / NR_);
    float var = sq[tid] * (1.0f / NR_) - mu * mu;
    float sd  = sqrtf(var + 1e-5f);
    smean[tid] = mu; sstd[tid] = sd;
    meanb[b * T_ + tid] = mu; stdb[b * T_ + tid] = sd;
  }
  __syncthreads();
  for (int e = tid; e < NR_ * T_; e += blockDim.x) {
    int t = e % T_;
    float v = x[(size_t)b * NR_ * T_ + e];
    xn[(size_t)b * NR_ * T_ + e] = (v - smean[t]) / sstd[t] * rw[t] + rb[t];
  }
}

// --------------------------- GAT prep: Wh = x@W (f16 transposed), f, g
template <int K, int F, int FP>
__global__ void k_gat_prep(const float* __restrict__ x, const float* __restrict__ W,
                           const float* __restrict__ a, _Float16* __restrict__ WhT,
                           float* __restrict__ fv, float* __restrict__ gv) {
  __shared__ float sW[K * F];
  __shared__ float sa[2 * F];
  for (int i = threadIdx.x; i < K * F; i += blockDim.x) sW[i] = W[i];
  for (int i = threadIdx.x; i < 2 * F; i += blockDim.x) sa[i] = a[i];
  __syncthreads();
  const int r = blockIdx.x * blockDim.x + threadIdx.x;
  if (r >= M_) return;
  float acc[F];
#pragma unroll
  for (int j = 0; j < F; ++j) acc[j] = 0.f;
  for (int k = 0; k < K; ++k) {
    float xv = x[(size_t)r * K + k];
#pragma unroll
    for (int j = 0; j < F; ++j) acc[j] += xv * sW[k * F + j];
  }
  float f = 0.f, g = 0.f;
#pragma unroll
  for (int j = 0; j < F; ++j) { f += acc[j] * sa[j]; g += acc[j] * sa[F + j]; }
  fv[r] = f; gv[r] = g;
#pragma unroll
  for (int j = 0; j < F; ++j) WhT[(size_t)j * M_ + r] = (_Float16)acc[j];
#pragma unroll
  for (int j = F; j < FP; ++j) WhT[(size_t)j * M_ + r] = (_Float16)0.f;
}

// ---------------------------------------------------------- max_j g[j]
__global__ void k_gmax(const float* __restrict__ g, float* __restrict__ gmax) {
  __shared__ float red[256];
  float m = -3.4e38f;
  for (int i = threadIdx.x; i < M_; i += 256) m = fmaxf(m, g[i]);
  red[threadIdx.x] = m;
  __syncthreads();
  for (int s = 128; s > 0; s >>= 1) {
    if (threadIdx.x < s) red[threadIdx.x] = fmaxf(red[threadIdx.x], red[threadIdx.x + s]);
    __syncthreads();
  }
  if (threadIdx.x == 0) *gmax = red[0];
}

// ----------------- softmax stats: m_i = lrelu(f_i+gmax), s_i = sum exp
__global__ void k_stats(const float* __restrict__ f, const float* __restrict__ g,
                        const float* __restrict__ gmax, float* __restrict__ mrow,
                        float* __restrict__ isrow) {
  const int wv = threadIdx.x >> 5, lane = threadIdx.x & 31;
  const int i = blockIdx.x * 8 + wv;
  const float fi = f[i];
  const float m = lrelu(fi + *gmax);  // lrelu monotone => this is row max
  float s = 0.f;
  for (int j = lane; j < M_; j += 32) s += __expf(lrelu(fi + g[j]) - m);
#pragma unroll
  for (int off = 16; off; off >>= 1) s += __shfl_xor(s, off, 32);
  if (lane == 0) { mrow[i] = m; isrow[i] = 1.f / s; }
}

// --------- WMMA attention aggregation: out = softmax(e) @ Wh  (+elu)
// One wave per 16 output rows; A-tile (attention probs) built in registers;
// g chunks double-buffered in LDS via async global->LDS copies.
template <int NT>
__global__ void __launch_bounds__(32)
k_attn_agg(const float* __restrict__ f, const float* __restrict__ g,
           const float* __restrict__ mrow, const float* __restrict__ isrow,
           const _Float16* __restrict__ WhT, float* __restrict__ out,
           int Fv, int col_off, int ostride, int do_elu) {
  __shared__ float gs[2][32];
  const int L = threadIdx.x;
  const int half = L >> 4;      // K-half selector for A/B layouts
  const int rA = L & 15;        // A row / B,C,D column
  const int i0 = blockIdx.x * 16;
  const float fi = f[i0 + rA], mi = mrow[i0 + rA], isi = isrow[i0 + rA];
  const unsigned lo0 = lds_off(&gs[0][L]);
  const unsigned lo1 = lds_off(&gs[1][L]);
  v8f c[NT] = {};
  ASYNC_LDS_B32(lo0, g + L);  // prefetch chunk 0
  for (int k0 = 0; k0 < M_; k0 += 32) {
    const int buf = (k0 >> 5) & 1;
    if (k0 + 32 < M_) {
      ASYNC_LDS_B32(buf ? lo0 : lo1, g + k0 + 32 + L);  // prefetch next chunk
      WAIT_ASYNC(0x1);  // async loads complete in order -> current chunk landed
      __builtin_prefetch(WhT + (size_t)rA * M_ + k0 + 256, 0, 1);
    } else {
      WAIT_ASYNC(0x0);
    }
    __syncthreads();
    v16h a;
#pragma unroll
    for (int e = 0; e < 16; ++e) {
      // A layout: element e of lane L -> K = e + 16*(L>=16)
      int ke = e + 16 * half;
      float ev = lrelu(fi + gs[buf][ke]);
      a[e] = (_Float16)(__expf(ev - mi) * isi);
    }
#pragma unroll
    for (int nt = 0; nt < NT; ++nt) {
      // B layout: lane L holds column (L&15), K = e + 16*(L>=16) -> contiguous
      const _Float16* bp = WhT + (size_t)(nt * 16 + rA) * M_ + k0 + 16 * half;
      v16h bt = *(const v16h*)bp;
      c[nt] = wmma_f16(a, bt, c[nt]);
    }
    __syncthreads();  // DS reads done before async engine refills this buffer
  }
#pragma unroll
  for (int nt = 0; nt < NT; ++nt) {
    int ncol = nt * 16 + rA;
    if (ncol < Fv) {
#pragma unroll
      for (int rv = 0; rv < 8; ++rv) {
        int orow = i0 + rv + 8 * half;  // C layout: VGPR rv, lanes>=16 -> M+8
        float v = c[nt][rv];
        if (do_elu) v = v > 0.f ? v : __expf(v) - 1.f;
        out[(size_t)orow * ostride + col_off + ncol] = v;
      }
    }
  }
}

// ---------------------------------------------------- z = mu + n*exp(.5mu)
__global__ void k_zsample(const float* __restrict__ h, float* __restrict__ z, int n) {
  int i = blockIdx.x * blockDim.x + threadIdx.x;
  if (i >= n) return;
  float m = h[i];
  z[i] = m + hash_gauss(0x5150a7u ^ (unsigned)i) * __expf(0.5f * m);
}

// ------------------------------- pack h2 (8x6144) into padded f16 A (16x6144)
__global__ void k_pack_a16(const float* __restrict__ h2, _Float16* __restrict__ A) {
  int i = blockIdx.x * blockDim.x + threadIdx.x;
  if (i >= 16 * 6144) return;
  int m = i / 6144, k = i % 6144;
  A[i] = (m < 8) ? (_Float16)h2[m * 6144 + k] : (_Float16)0.f;
}

// --------------- fc1: (16pad x 6144) @ (6144 x 12288) WMMA
// 302 MB weight stream double-buffered through LDS with async b128 copies.
__global__ void __launch_bounds__(32)
k_fc1_wmma(const _Float16* __restrict__ A, const float* __restrict__ W,
           const float* __restrict__ bias, float* __restrict__ out) {
  constexpr int N = 12288, K = 6144;
  __shared__ float sw[2][32 * 64];  // raw fp32 tiles [k_local][n_local]
  const int L = threadIdx.x;
  const int half = L >> 4, rA = L & 15;
  const int n0 = blockIdx.x * 64;
  v8f c[4] = {};
  {  // prefetch tile 0: lane L copies row k=L (64 floats = 16 async b128)
    const float* gp = W + (size_t)L * N + n0;
    const unsigned lo = lds_off(&sw[0][L * 64]);
#pragma unroll
    for (int q = 0; q < 16; ++q) ASYNC_LDS_B128(lo + 16u * q, gp + 4 * q);
  }
  for (int k0 = 0; k0 < K; k0 += 32) {
    const int buf = (k0 >> 5) & 1;
    if (k0 + 32 < K) {
      const float* gp = W + (size_t)(k0 + 32 + L) * N + n0;
      const unsigned lo = lds_off(&sw[buf ^ 1][L * 64]);
#pragma unroll
      for (int q = 0; q < 16; ++q) ASYNC_LDS_B128(lo + 16u * q, gp + 4 * q);
      WAIT_ASYNC(0x10);  // <=16 outstanding -> the 16 copies of tile k0 landed
    } else {
      WAIT_ASYNC(0x0);
    }
    __syncthreads();
    // A tile: lane row rA; elements 0..7 -> K=k0+8*half+e, 8..15 -> +16
    const _Float16* ap = A + (size_t)rA * K + k0 + 8 * half;
    v8h lo8 = *(const v8h*)ap;
    v8h hi8 = *(const v8h*)(ap + 16);
    v16h a;
#pragma unroll
    for (int e = 0; e < 8; ++e) { a[e] = lo8[e]; a[e + 8] = hi8[e]; }
#pragma unroll
    for (int nt = 0; nt < 4; ++nt) {
      v16h bt;
#pragma unroll
      for (int e = 0; e < 16; ++e)  // B: col nt*16+rA, K = e + 8*half (+8 for e>=8)
        bt[e] = (_Float16)sw[buf][(e + 8 * half + (e & 8)) * 64 + nt * 16 + rA];
      c[nt] = wmma_f16(a, bt, c[nt]);
    }
    __syncthreads();  // DS reads done before async engine refills this buffer
  }
  if (half == 0) {  // valid output rows M=0..7 live in lanes 0..15
#pragma unroll
    for (int nt = 0; nt < 4; ++nt) {
      int n = n0 + nt * 16 + rA;
#pragma unroll
      for (int rv = 0; rv < 8; ++rv)
        out[(size_t)rv * N + n] = c[nt][rv] + bias[n];
    }
  }
}

// ---------- forward diffusion + 10-step reverse denoise + RevIN denorm
__global__ void k_diffuse(const float* __restrict__ X0, const float* __restrict__ W1,
                          const float* __restrict__ b1, const float* __restrict__ W2,
                          const float* __restrict__ b2, const float* __restrict__ W3,
                          const float* __restrict__ b3, const float* __restrict__ rw,
                          const float* __restrict__ rb, const float* __restrict__ meanb,
                          const float* __restrict__ stdb, float* __restrict__ out) {
  const int r = blockIdx.x * blockDim.x + threadIdx.x;  // row over B*NL
  if (r >= B_ * NL_) return;
  float X[T_];
#pragma unroll
  for (int t = 0; t < T_; ++t) X[t] = X0[(size_t)r * T_ + t];
  // forward diffusion
  for (int s = 0; s < 10; ++s) {
    float al = 0.1f + 0.8f * (float)s / 9.0f;
    float sa = sqrtf(al), sb = sqrtf(1.f - al);
#pragma unroll
    for (int t = 0; t < T_; ++t)
      X[t] = sa * X[t] + sb * hash_gauss((unsigned)(r * T_ + t) * 41u + (unsigned)s * 9973u + 7u);
  }
  // reverse diffusion: per-row 12->50->50->12 MLP
  for (int s = 9; s >= 0; --s) {
    float h1[50], h2v[50];
    for (int j = 0; j < 50; ++j) {
      float acc = b1[j];
#pragma unroll
      for (int t = 0; t < T_; ++t) acc += X[t] * W1[t * 50 + j];
      h1[j] = fmaxf(acc, 0.f);
    }
    for (int j = 0; j < 50; ++j) {
      float acc = b2[j];
      for (int t = 0; t < 50; ++t) acc += h1[t] * W2[t * 50 + j];
      h2v[j] = fmaxf(acc, 0.f);
    }
#pragma unroll
    for (int t = 0; t < T_; ++t) {
      float o = b3[t];
      for (int j = 0; j < 50; ++j) o += h2v[j] * W3[j * T_ + t];
      float sig = 1.f / (1.f + __expf(-o));
      X[t] = o + sig * hash_gauss((unsigned)(r * T_ + t) * 2654435761u + (unsigned)s * 193939u + 0xbeefu);
    }
  }
  // RevIN denorm
  const int b = r / NL_;
#pragma unroll
  for (int t = 0; t < T_; ++t)
    out[(size_t)r * T_ + t] =
        (X[t] - rb[t]) / (rw[t] + 1e-10f) * stdb[b * T_ + t] + meanb[b * T_ + t];
}

// =========================================================== host side
extern "C" void kernel_launch(void* const* d_in, const int* in_sizes, int n_in,
                              void* d_out, int out_size, void* d_ws, size_t ws_size,
                              hipStream_t stream) {
  (void)in_sizes; (void)n_in; (void)out_size; (void)ws_size;
  const float* x_R   = (const float*)d_in[0];
  const float* encWh = (const float*)d_in[1];
  const float* encah = (const float*)d_in[2];
  const float* encWo = (const float*)d_in[3];
  const float* encao = (const float*)d_in[4];
  const float* decWh = (const float*)d_in[5];
  const float* decah = (const float*)d_in[6];
  const float* decWo = (const float*)d_in[7];
  const float* decao = (const float*)d_in[8];
  const float* fc1W  = (const float*)d_in[9];
  const float* fc1b  = (const float*)d_in[10];
  const float* dnW1  = (const float*)d_in[11];
  const float* dnb1  = (const float*)d_in[12];
  const float* dnW2  = (const float*)d_in[13];
  const float* dnb2  = (const float*)d_in[14];
  const float* dnW3  = (const float*)d_in[15];
  const float* dnb3  = (const float*)d_in[16];
  const float* revw  = (const float*)d_in[17];
  const float* revb  = (const float*)d_in[18];

  char* ws = (char*)d_ws;
  size_t off = 0;
  auto carve = [&](size_t bytes) -> void* {
    void* p = ws + off;
    off += (bytes + 255) & ~(size_t)255;
    return p;
  };
  float*     meanb = (float*)carve(B_ * T_ * 4);
  float*     stdb  = (float*)carve(B_ * T_ * 4);
  float*     xn    = (float*)carve((size_t)M_ * T_ * 4);
  float*     fbuf  = (float*)carve(M_ * 4);
  float*     gbuf  = (float*)carve(M_ * 4);
  float*     gmaxb = (float*)carve(4);
  float*     mrow  = (float*)carve(M_ * 4);
  float*     isrow = (float*)carve(M_ * 4);
  _Float16*  WhT   = (_Float16*)carve((size_t)64 * M_ * 2);
  float*     xcat  = (float*)carve((size_t)M_ * 150 * 4);
  float*     zmean = (float*)carve((size_t)M_ * 30 * 4);
  float*     zbuf  = (float*)carve((size_t)M_ * 30 * 4);
  float*     h2b   = (float*)carve((size_t)M_ * T_ * 4);
  _Float16*  A16   = (_Float16*)carve((size_t)16 * 6144 * 2);
  float*     Xbuf  = (float*)carve((size_t)8 * 12288 * 4);

  auto gat_layer = [&](const float* xin, const float* Wp, const float* ap,
                       int Kv, int Fv, float* outp, int col_off, int ostride,
                       int do_elu) {
    if (Kv == 12)
      k_gat_prep<12, 50, 64><<<M_ / 256, 256, 0, stream>>>(xin, Wp, ap, WhT, fbuf, gbuf);
    else if (Kv == 30)
      k_gat_prep<30, 50, 64><<<M_ / 256, 256, 0, stream>>>(xin, Wp, ap, WhT, fbuf, gbuf);
    else if (Fv == 30)
      k_gat_prep<150, 30, 32><<<M_ / 256, 256, 0, stream>>>(xin, Wp, ap, WhT, fbuf, gbuf);
    else
      k_gat_prep<150, 12, 16><<<M_ / 256, 256, 0, stream>>>(xin, Wp, ap, WhT, fbuf, gbuf);
    k_gmax<<<1, 256, 0, stream>>>(gbuf, gmaxb);
    k_stats<<<M_ / 8, 256, 0, stream>>>(fbuf, gbuf, gmaxb, mrow, isrow);
    int NT = (Fv + 15) / 16;
    if (NT == 4)
      k_attn_agg<4><<<M_ / 16, 32, 0, stream>>>(fbuf, gbuf, mrow, isrow, WhT, outp,
                                                Fv, col_off, ostride, do_elu);
    else if (NT == 2)
      k_attn_agg<2><<<M_ / 16, 32, 0, stream>>>(fbuf, gbuf, mrow, isrow, WhT, outp,
                                                Fv, col_off, ostride, do_elu);
    else
      k_attn_agg<1><<<M_ / 16, 32, 0, stream>>>(fbuf, gbuf, mrow, isrow, WhT, outp,
                                                Fv, col_off, ostride, do_elu);
  };

  // 1) RevIN normalize
  k_revin<<<B_, 256, 0, stream>>>(x_R, revw, revb, meanb, stdb, xn);
  // 2) encoder: 3 GAT heads (T->50, elu, concat) then output GAT (150->30)
  for (int h = 0; h < 3; ++h)
    gat_layer(xn, encWh + h * 12 * 50, encah + h * 100, 12, 50, xcat, h * 50, 150, 1);
  gat_layer(xcat, encWo, encao, 150, 30, zmean, 0, 30, 0);
  // 3) reparameterize
  k_zsample<<<(M_ * 30 + 255) / 256, 256, 0, stream>>>(zmean, zbuf, M_ * 30);
  // 4) decoder: 3 GAT heads (30->50) then output GAT (150->12)
  for (int h = 0; h < 3; ++h)
    gat_layer(zbuf, decWh + h * 30 * 50, decah + h * 100, 30, 50, xcat, h * 50, 150, 1);
  gat_layer(xcat, decWo, decao, 150, 12, h2b, 0, 12, 0);
  // 5) fc1 GEMM (8x6144 @ 6144x12288) via WMMA + async double-buffered weights
  k_pack_a16<<<(16 * 6144 + 255) / 256, 256, 0, stream>>>(h2b, A16);
  k_fc1_wmma<<<12288 / 64, 32, 0, stream>>>(A16, fc1W, fc1b, Xbuf);
  // 6) diffusion forward + reverse + denorm, straight to output
  k_diffuse<<<(B_ * NL_ + 255) / 256, 256, 0, stream>>>(
      Xbuf, dnW1, dnb1, dnW2, dnb2, dnW3, dnb3, revw, revb, meanb, stdb,
      (float*)d_out);
}
